// PreQ_90692529422531
// MI455X (gfx1250) — matrix operand order repaired and checked by reading, not compiled, not run on hardware
//
#include <hip/hip_runtime.h>
#include <hip/hip_bf16.h>

typedef _Float16 v16h __attribute__((ext_vector_type(16)));
typedef _Float16 h8   __attribute__((ext_vector_type(8)));
typedef _Float16 h4   __attribute__((ext_vector_type(4)));
typedef float    v8f  __attribute__((ext_vector_type(8)));

#define NWAVES 8
#define TPB 256

// Padded layer shapes (K padded to x32, N padded to x16):
// L1: I=300 Kp=320  O=200 Np=208
// L2: I=200 Kp=224  O=100 Np=112
// L3: I=100 Kp=128  O=50  Np=64
// L4: I=50  Kp=64   O=10  Np=16

// ---------------------------------------------------------------------------
// Weight prep: fp32 W[O][I] -> fp16 fragment-ordered blocks of 512 halves.
// Block (ks, nt): lane ln holds column n = nt*16 + (ln&15),
//   halves j=0..15 hold K = ks*32 + (ln>>4)*16 + j  (contiguous k order, same as A).
// ---------------------------------------------------------------------------
__global__ void prep_weights(const float* __restrict__ W, _Float16* __restrict__ out,
                             int O, int I, int Kp, int Np) {
  int idx = blockIdx.x * blockDim.x + threadIdx.x;
  int total = Kp * Np;
  if (idx >= total) return;
  int blk = idx >> 9;
  int rr  = idx & 511;
  int lane = rr >> 4;
  int j    = rr & 15;
  int ntiles = Np >> 4;
  int ks = blk / ntiles;
  int nt = blk - ks * ntiles;
  int n = nt * 16 + (lane & 15);
  int k = ks * 32 + ((lane >> 4) << 4) + j;
  float v = (n < O && k < I) ? W[n * I + k] : 0.0f;
  out[idx] = (_Float16)v;
}

// ---------------------------------------------------------------------------
// A fragment (16x32 fp16) from LDS, contiguous k order:
// lanes 0-15: M=lane, K = kbase+{0..15}; lanes 16-31: M=lane-16,
// K = kbase+{16..31}.
// ---------------------------------------------------------------------------
__device__ __forceinline__ v16h load_a_frag(const _Float16* acts, int stride,
                                            int mbase, int ks, int lane) {
  int row = mbase + (lane & 15);
  int off = (lane >> 4) << 4;  // 0 or 16
  const _Float16* p = acts + row * stride + ks * 32 + off;
  h8 lo = *(const h8*)p;
  h8 hi = *(const h8*)(p + 8);
  v16h a;
#pragma unroll
  for (int i = 0; i < 8; ++i) { a[i] = lo[i]; a[i + 8] = hi[i]; }
  return a;
}

// One MLP layer over a 32x16 virtual-row tile; each wave owns N-tiles
// nt = wave, wave+8, ...  Two WMMAs per K-step (M-tiles 0..15 and 16..31).
__device__ __forceinline__ void mlp_layer(const _Float16* __restrict__ Wf,
                                          const float* __restrict__ bias,
                                          const _Float16* actsIn, int Kp,
                                          _Float16* actsOut, int ostride,
                                          float* fout,
                                          int Np, int O, bool do_relu,
                                          int wave, int lane) {
  const int ntiles = Np >> 4;
  const int ksteps = Kp >> 5;
  for (int nt = wave; nt < ntiles; nt += NWAVES) {
    const int n = nt * 16 + (lane & 15);
    const float bv = (n < O) ? bias[n] : 0.0f;   // C layout: lane's N = lane&15
    v8f acc0 = {bv, bv, bv, bv, bv, bv, bv, bv};
    v8f acc1 = acc0;
    const _Float16* wp = Wf + ((size_t)nt << 9) + (lane << 4);
    for (int ks = 0; ks < ksteps; ++ks) {
      v16h a0 = load_a_frag(actsIn, Kp, 0, ks, lane);
      v16h a1 = load_a_frag(actsIn, Kp, 16, ks, lane);
      v16h b  = *(const v16h*)(wp + (((size_t)ks * ntiles) << 9));
      if (ks + 1 < ksteps)
        __builtin_prefetch(wp + (((size_t)(ks + 1) * ntiles) << 9), 0, 3);
      acc0 = __builtin_amdgcn_wmma_f32_16x16x32_f16(false, a0, false, b,
                                                    (short)0, acc0, false, false);
      acc1 = __builtin_amdgcn_wmma_f32_16x16x32_f16(false, a1, false, b,
                                                    (short)0, acc1, false, false);
    }
#pragma unroll
    for (int r = 0; r < 8; ++r) {
      int row = r + ((lane >> 4) << 3);  // C layout: M = r (+8 for hi lanes)
      float f0 = acc0[r], f1 = acc1[r];
      if (do_relu) { f0 = fmaxf(f0, 0.0f); f1 = fmaxf(f1, 0.0f); }
      if (fout) {
        fout[row * 16 + (lane & 15)]        = f0;
        fout[(row + 16) * 16 + (lane & 15)] = f1;
      } else {
        actsOut[row * ostride + n]        = (_Float16)f0;
        actsOut[(row + 16) * ostride + n] = (_Float16)f1;
      }
    }
  }
}

// ---------------------------------------------------------------------------
// 4x4 complex quantum circuit (scalar, negligible FLOPs)
// ---------------------------------------------------------------------------
struct c32 { float re, im; };
__device__ __forceinline__ c32 cmul(c32 a, c32 b) {
  return {a.re * b.re - a.im * b.im, a.re * b.im + a.im * b.re};
}
__device__ __forceinline__ c32 cadd(c32 a, c32 b) { return {a.re + b.re, a.im + b.im}; }
__device__ __forceinline__ c32 cscale(c32 a, float s) { return {a.re * s, a.im * s}; }

__device__ void qstate(float p1, float p2, float p3, float p4, c32 U[4][4]) {
  float s1, c1; __sincosf(0.5f * p1, &s1, &c1);
  float s2, c2; __sincosf(0.5f * p2, &s2, &c2);
  float s3, c3; __sincosf(0.5f * p3, &s3, &c3);
  float s4, c4; __sincosf(0.5f * p4, &s4, &c4);
  c32 rx1[2][2] = {{{c1, 0.f}, {0.f, -s1}}, {{0.f, -s1}, {c1, 0.f}}};
  float ry2[2][2] = {{c2, -s2}, {s2, c2}};
  float ry3[2][2] = {{c3, -s3}, {s3, c3}};
  c32 F[4][4]; float S[4][4];
#pragma unroll
  for (int i = 0; i < 2; ++i)
    for (int k = 0; k < 2; ++k)
      for (int j = 0; j < 2; ++j)
        for (int l = 0; l < 2; ++l) {
          F[2 * i + k][2 * j + l] = cmul(rx1[i][j], rx1[k][l]);
          S[2 * i + k][2 * j + l] = ry2[i][j] * ry3[k][l];
        }
  c32 rot[4][4];
#pragma unroll
  for (int r = 0; r < 4; ++r)
    for (int c = 0; c < 4; ++c) {
      c32 acc = {0.f, 0.f};
#pragma unroll
      for (int m = 0; m < 4; ++m) acc = cadd(acc, cscale(F[r][m], S[m][c]));
      rot[r][c] = acc;
    }
  // U = rot @ crx(p4); crx col0=e0, col2=e2, cols 1/3 mix via Rx(p4)
  c32 rxc[2][2] = {{{c4, 0.f}, {0.f, -s4}}, {{0.f, -s4}, {c4, 0.f}}};
#pragma unroll
  for (int r = 0; r < 4; ++r) {
    U[r][0] = rot[r][0];
    U[r][2] = rot[r][2];
    U[r][1] = cadd(cmul(rot[r][1], rxc[0][0]), cmul(rot[r][3], rxc[1][0]));
    U[r][3] = cadd(cmul(rot[r][1], rxc[0][1]), cmul(rot[r][3], rxc[1][1]));
  }
}

// ---------------------------------------------------------------------------
// Fused kernel: 16 batch rows per block (32 virtual rows), 8 waves.
// ---------------------------------------------------------------------------
__global__ __launch_bounds__(TPB)
void fused_mlp_fidelity(const float* __restrict__ x,
                        const _Float16* __restrict__ Wf1, const float* __restrict__ b1,
                        const _Float16* __restrict__ Wf2, const float* __restrict__ b2,
                        const _Float16* __restrict__ Wf3, const float* __restrict__ b3,
                        const _Float16* __restrict__ Wf4, const float* __restrict__ b4,
                        float* __restrict__ out) {
  __shared__ _Float16 buf0[32 * 320];
  __shared__ _Float16 buf1[32 * 224];
  __shared__ _Float16 buf2[32 * 128];
  __shared__ _Float16 buf3[32 * 64];
  __shared__ float    sOut[32 * 16];

  const int tid  = threadIdx.x;
  const int lane = tid & 31;
  const int wave = tid >> 5;
  const int mbase = blockIdx.x * 16;

  // Stage x: virtual rows 0..15 <- x[:, :300], rows 16..31 <- x[:, 300:600].
  // K-pad cols 300..319 zeroed (300 is a multiple of 4 so chunks don't straddle).
  for (int i = tid * 4; i < 32 * 320; i += TPB * 4) {
    int row = i / 320;
    int k = i - row * 320;
    h4 hv = {(_Float16)0.f, (_Float16)0.f, (_Float16)0.f, (_Float16)0.f};
    if (k < 300) {
      const float4 v = *(const float4*)(x + (size_t)(mbase + (row & 15)) * 600 +
                                        (row >> 4) * 300 + k);
      hv[0] = (_Float16)v.x; hv[1] = (_Float16)v.y;
      hv[2] = (_Float16)v.z; hv[3] = (_Float16)v.w;
    }
    *(h4*)(buf0 + i) = hv;
  }
  // Zero K-pad columns of intermediate buffers (buf1 cols 208..223, buf2 112..127).
  for (int i = tid; i < 32 * 16; i += TPB) {
    int r = i >> 4, c = i & 15;
    buf1[r * 224 + 208 + c] = (_Float16)0.f;
    buf2[r * 128 + 112 + c] = (_Float16)0.f;
  }
  __syncthreads();

  mlp_layer(Wf1, b1, buf0, 320, buf1, 224, nullptr, 208, 200, true,  wave, lane);
  __syncthreads();
  mlp_layer(Wf2, b2, buf1, 224, buf2, 128, nullptr, 112, 100, true,  wave, lane);
  __syncthreads();
  mlp_layer(Wf3, b3, buf2, 128, buf3,  64, nullptr,  64,  50, true,  wave, lane);
  __syncthreads();
  mlp_layer(Wf4, b4, buf3,  64, nullptr, 0, sOut,    16,  10, false, wave, lane);
  __syncthreads();

  if (tid < 16) {
    const float* l1 = sOut + tid * 16;
    const float* l2 = sOut + (tid + 16) * 16;
    c32 U1a[4][4], U1b[4][4], U2a[4][4], U2b[4][4];
    qstate(l1[1], l1[2], l1[3], l1[4], U1a);
    qstate(l1[6], l1[7], l1[8], l1[9], U1b);
    qstate(l2[1], l2[2], l2[3], l2[4], U2a);
    qstate(l2[6], l2[7], l2[8], l2[9], U2b);
    // v = U1b @ U1a[:,0]; w = U2b @ U2a[:,0]; inner = <w, v>
    c32 v[4], w[4];
#pragma unroll
    for (int r = 0; r < 4; ++r) {
      c32 av = {0.f, 0.f}, aw = {0.f, 0.f};
#pragma unroll
      for (int c = 0; c < 4; ++c) {
        av = cadd(av, cmul(U1b[r][c], U1a[c][0]));
        aw = cadd(aw, cmul(U2b[r][c], U2a[c][0]));
      }
      v[r] = av; w[r] = aw;
    }
    float re = 0.f, im = 0.f;
#pragma unroll
    for (int r = 0; r < 4; ++r) {
      re += w[r].re * v[r].re + w[r].im * v[r].im;
      im += w[r].re * v[r].im - w[r].im * v[r].re;
    }
    out[mbase + tid] = re * re + im * im;
  }
}

extern "C" void kernel_launch(void* const* d_in, const int* in_sizes, int n_in,
                              void* d_out, int out_size, void* d_ws, size_t ws_size,
                              hipStream_t stream) {
  const float* x  = (const float*)d_in[0];
  const float* W1 = (const float*)d_in[1];
  const float* b1 = (const float*)d_in[2];
  const float* W2 = (const float*)d_in[3];
  const float* b2 = (const float*)d_in[4];
  const float* W3 = (const float*)d_in[5];
  const float* b3 = (const float*)d_in[6];
  const float* W4 = (const float*)d_in[7];
  const float* b4 = (const float*)d_in[8];

  _Float16* ws  = (_Float16*)d_ws;
  _Float16* Wf1 = ws;                          // 320*208 = 66560 halves
  _Float16* Wf2 = ws + 66560;                  // 224*112 = 25088
  _Float16* Wf3 = ws + 66560 + 25088;          // 128*64  = 8192
  _Float16* Wf4 = ws + 66560 + 25088 + 8192;   // 64*16   = 1024

  prep_weights<<<(66560 + 255) / 256, 256, 0, stream>>>(W1, Wf1, 200, 300, 320, 208);
  prep_weights<<<(25088 + 255) / 256, 256, 0, stream>>>(W2, Wf2, 100, 200, 224, 112);
  prep_weights<<<( 8192 + 255) / 256, 256, 0, stream>>>(W3, Wf3,  50, 100, 128,  64);
  prep_weights<<<( 1024 + 255) / 256, 256, 0, stream>>>(W4, Wf4,  10,  50,  64,  16);

  const int Bn = in_sizes[0] / 600;  // 65536
  fused_mlp_fidelity<<<Bn / 16, TPB, 0, stream>>>(x, Wf1, b1, Wf2, b2, Wf3, b3,
                                                  Wf4, b4, (float*)d_out);
}
